// NetworkRateCNN_22179211117458
// MI455X (gfx1250) — matrix operand order repaired and checked by reading, not compiled, not run on hardware
//
#include <hip/hip_runtime.h>

// ---------------------------------------------------------------------------
// SLAYER-style spiking CNN forward pass for gfx1250 (MI455X).
//   - psp temporal filter computed as exact 2nd-order IIR (O(T) per site)
//   - all linear layers as implicit GEMM via V_WMMA_F32_16X16X32_F16
//   - spikes {0,1} exact in f16; f32 accumulation inside WMMA, pre-activations
//     rounded once to f16 on store (halves the dominant memory traffic)
//   - zero-halo LDS tiles -> branch-free B-fragment gathers
//   - conv2 stages activations with GLOBAL_LOAD_ASYNC_TO_LDS_B128 (ASYNCcnt)
// ---------------------------------------------------------------------------

typedef __attribute__((ext_vector_type(16))) _Float16 v16h;
typedef __attribute__((ext_vector_type(8)))  float    v8f;

#define T_SIM 100
#define ALPHA 0.9048374180359595f   /* exp(-1/tauSr), tauSr = 10 */
#define PSPC  0.2718281828459045f   /* e / tauSr                  */

__device__ __forceinline__ v8f wmma_f16(v16h a, v16h b, v8f c) {
  // D = A(16x32 f16) * B(32x16 f16) + C(16x16 f32)
  return __builtin_amdgcn_wmma_f32_16x16x32_f16(false, a, false, b,
                                                (short)0, c, false, false);
}

// ---------------------------------------------------------------------------
// Weight packing into per-lane WMMA A-fragment layout.
// A-fragment as packed here (B uses same K order): lane L holds row M = L%16; halves j=0..15 map
// to K = j + (L<16 ? 0 : 16).  Packed so each lane's runtime
// load is one contiguous 32-byte v16h: idx = ((mtile*Ksteps+ks)*32+L)*16+j.
// mode 0: W[m*Klog + k]     (dense / conv1; zero-pad m>=O or k>=Klog)
// mode 1: conv2 reorder, k = q*16+i, q = kh*5+kw, i = input channel:
//         W[(m*16+i)*25 + q] (zero-pad q>=25)
// ---------------------------------------------------------------------------
__global__ void pack_a_kernel(const float* __restrict__ W, _Float16* __restrict__ A,
                              int O, int Klog, int Mtiles, int Ksteps, int mode) {
  int tid = blockIdx.x * 256 + threadIdx.x;
  int total = Mtiles * Ksteps * 512;
  if (tid >= total) return;
  int j    = tid & 15;
  int lane = (tid >> 4) & 31;
  int frag = tid >> 9;              // mtile*Ksteps + ks
  int ks    = frag % Ksteps;
  int mtile = frag / Ksteps;
  int m = mtile * 16 + (lane & 15);
  int klocal = j + ((lane < 16) ? 0 : 16);
  int k = ks * 32 + klocal;
  float v = 0.0f;
  if (mode == 0) {
    if (m < O && k < Klog) v = W[m * Klog + k];
  } else {
    int q = k >> 4, i = k & 15;
    if (q < 25 && m < O) v = W[(m * 16 + i) * 25 + q];
  }
  A[tid] = (_Float16)v;
}

// ---------------------------------------------------------------------------
// Rate encoding: spk0[(b*T+t)*784 + hw] = (rand_u[b,hw,t] < img[b,hw])
// ---------------------------------------------------------------------------
__global__ void encode_kernel(const float* __restrict__ img,
                              const float* __restrict__ ru,
                              _Float16* __restrict__ spk) {
  int tid = blockIdx.x * 256 + threadIdx.x;
  if (tid >= 32 * 784 * T_SIM) return;
  int hw = tid % 784;
  int bt = tid / 784;
  int t = bt % T_SIM;
  int b = bt / T_SIM;
  float r = ru[(b * 784 + hw) * T_SIM + t];
  float p = img[b * 784 + hw];
  spk[tid] = (_Float16)((r < p) ? 1.0f : 0.0f);
}

// ---------------------------------------------------------------------------
// conv1: 1->16 ch, 5x5 pad 2 on 28x28, per (b,t).  M=16, K=25(->32), N=784.
// Spike plane staged into a zeroed 32x32 padded LDS tile at (+2,+2): every
// tap (h0+kh, w0+kw) is unconditionally in-range; K-pad slots (k>=25) read
// offset 31*32 from pbase (A is zero there).  Per-chunk tap offsets are compile-time constants
// selected by lane<16 -> branch-free gathers.
// u1 column-major f16: u1[(bt*784 + col)*16 + row].
// ---------------------------------------------------------------------------
__global__ void __launch_bounds__(128)
conv1_kernel(const _Float16* __restrict__ spk0, const _Float16* __restrict__ Apk,
             _Float16* __restrict__ u1) {
  __shared__ __align__(32) _Float16 lds[32 * 32];
  int bt  = blockIdx.x;
  int tid = threadIdx.x;

  unsigned* zp = (unsigned*)lds;
  for (int i = tid; i < 512; i += 128) zp[i] = 0u;
  __syncthreads();
  const unsigned* src = (const unsigned*)(spk0 + (long)bt * 784);
  for (int i = tid; i < 392; i += 128) {
    int r = i / 14, cw = i % 14;                 // 14 dwords per 28-px row
    zp[(r + 2) * 16 + 1 + cw] = src[i];          // interior at (+2,+2)
  }
  __syncthreads();

  int lane = tid & 31;
  int wv   = tid >> 5;
  bool lo  = (lane < 16);
  v16h a = *(const v16h*)(Apk + lane * 16);
  int colm = lane & 15;
  int roff = lo ? 0 : 8;

  for (int ntile = wv; ntile < 49; ntile += 4) {
    int col = ntile * 16 + colm;                 // 49*16 == 784, in range
    int h0 = col / 28, w0 = col % 28;
    int pbase = h0 * 32 + w0;
    v16h bfr;
#pragma unroll
    for (int jj = 0; jj < 16; ++jj) {
      const int kA = jj;                          // chunk for lanes 0-15
      const int kB = jj + 16;                     // chunk for lanes 16-31
      const int oA = (kA / 5) * 32 + (kA % 5);
      const int oB = (kB < 25) ? ((kB / 5) * 32 + (kB % 5)) : (31 * 32);
      bfr[jj] = lds[pbase + (lo ? oA : oB)];
    }
    v8f acc = {};
    acc = wmma_f16(a, bfr, acc);
    _Float16* out = u1 + ((long)bt * 784 + col) * 16 + roff;
#pragma unroll
    for (int r = 0; r < 8; ++r) out[r] = (_Float16)acc[r];
  }
}

// ---------------------------------------------------------------------------
// Fused SRM-psp + threshold, exact 2nd-order IIR (f16 in, f16 spikes out).
// bt_major=1: u laid out [b][t][slice]; base = b*T*slice + r, stride = slice.
// bt_major=0: u laid out [t][total];   base = tid,           stride = total.
// ---------------------------------------------------------------------------
__global__ void psp_spike_kernel(const _Float16* __restrict__ u,
                                 _Float16* __restrict__ s,
                                 int total, int slice, int bt_major) {
  int tid = blockIdx.x * 256 + threadIdx.x;
  if (tid >= total) return;
  long base, stride;
  if (bt_major) {
    int b = tid / slice, r = tid % slice;
    base = (long)b * T_SIM * slice + r;
    stride = slice;
  } else {
    base = tid;
    stride = total;
  }
  float p = 0.0f, q = 0.0f;
  for (int t = 0; t < T_SIM; ++t) {
    float x = (float)u[base + (long)t * stride];
    p = ALPHA * p + x;
    q = ALPHA * q + p;
    float y = PSPC * (q - p);
    s[base + (long)t * stride] = (_Float16)((y >= 1.0f) ? 1.0f : 0.0f);
  }
}

// ---------------------------------------------------------------------------
// pool(2x2 sum *1.1) + psp + spike, layer 2.
// in  s1[(b*T+t)*784*16 + site*16 + c]   (16 ch, 28x28)
// out s2[(b*T+t)*196*16 + site2*16 + c]  (16 ch, 14x14) -> conv2 LDS layout
// ---------------------------------------------------------------------------
__global__ void pool_psp1_kernel(const _Float16* __restrict__ s1,
                                 _Float16* __restrict__ s2) {
  int tid = blockIdx.x * 256 + threadIdx.x;
  if (tid >= 32 * 196 * 16) return;
  int c    = tid & 15;
  int rest = tid >> 4;
  int site = rest % 196;
  int b    = rest / 196;
  int h2 = site / 14, w2 = site % 14;
  int s00 = (h2 * 2) * 28 + w2 * 2;
  float p = 0.0f, q = 0.0f;
  for (int t = 0; t < T_SIM; ++t) {
    long base = ((long)(b * T_SIM + t) * 784) * 16 + c;
    float x = (float)s1[base + s00 * 16] + (float)s1[base + (s00 + 1) * 16]
            + (float)s1[base + (s00 + 28) * 16] + (float)s1[base + (s00 + 29) * 16];
    x *= 1.1f;
    p = ALPHA * p + x;
    q = ALPHA * q + p;
    float y = PSPC * (q - p);
    s2[((long)(b * T_SIM + t) * 196 + site) * 16 + c] =
        (_Float16)((y >= 1.0f) ? 1.0f : 0.0f);
  }
}

// ---------------------------------------------------------------------------
// conv2: 16->32 ch, 5x5 pad 2 on 14x14.  M=32 (2 tiles), K=400->416 (13
// steps), N=196->208 (13 tiles).  K ordered (kh,kw)-outer / channel-inner so
// each lane's 16 B values = one contiguous [site][16ch] vector (2x ds b128).
// The 14x14x16 activation tile is async-DMA'd (GLOBAL_LOAD_ASYNC_TO_LDS_B128)
// into a zeroed 19x18x16 padded LDS tile: zero halo absorbs the spatial
// bounds, the K-pad chunk (q==25) and the N-pad columns (col>=196), so the
// inner loop is completely branch-free.
// u2 column-major f16: u2[(bt*196 + col)*32 + row].
// ---------------------------------------------------------------------------
__global__ void __launch_bounds__(256)
conv2_kernel(const _Float16* __restrict__ s2, const _Float16* __restrict__ Apk,
             _Float16* __restrict__ u2) {
  __shared__ __align__(32) _Float16 lds[19 * 18 * 16];   // rows -2..16, cols -2..15
  int bt  = blockIdx.x;
  int tid = threadIdx.x;

  // Zero the whole padded tile (halo must be 0), then async-load interior.
  unsigned* zp = (unsigned*)lds;
  for (int i = tid; i < (19 * 18 * 16) / 2; i += 256) zp[i] = 0u;
  __syncthreads();
  {
    const _Float16* src = s2 + (long)bt * 3136;          // [196 sites][16 ch]
    for (int i = tid; i < 392; i += 256) {               // 392 x 16B chunks
      int site = i >> 1;
      int h = site / 14, w = site % 14;
      unsigned ldsbyte =
          (unsigned)(((((h + 2) * 18 + (w + 2)) * 16) + (i & 1) * 8) * 2);
      unsigned long long ga = (unsigned long long)(const void*)(src + i * 8);
      // LDS[ldsbyte..+15] = MEM[ga..+15], tracked by ASYNCcnt.
      asm volatile("global_load_async_to_lds_b128 %0, %1, off"
                   :: "v"(ldsbyte), "v"(ga) : "memory");
    }
  }
  asm volatile("s_wait_asynccnt 0" ::: "memory");
  __syncthreads();

  int lane   = tid & 31;
  int wv     = tid >> 5;
  int koff16 = (lane < 16) ? 0 : 1;  // which 16-chunk of the 32-K step
  int colm   = lane & 15;
  int roff   = (lane < 16) ? 0 : 8;

  for (int tile = wv; tile < 26; tile += 8) {
    int mtile = tile & 1;
    int ntile = tile >> 1;
    int col = ntile * 16 + colm;
    int h0 = col / 14, w0 = col % 14;       // col>=196 -> h0==14; not stored below
    int pbase = (h0 * 18 + w0) * 16;
    v8f acc = {};
#pragma unroll
    for (int ks = 0; ks < 13; ++ks) {
      v16h a = *(const v16h*)(Apk + ((mtile * 13 + ks) * 32 + lane) * 16);
      int qq = ks * 2 + koff16;             // kh*5+kw (q==25: A weights are zero)
      int off = ((qq / 5) * 18 + (qq % 5)) * 16;   // folds after unroll
      v16h bfr = *(const v16h*)(lds + pbase + off);
      acc = wmma_f16(a, bfr, acc);
    }
    if (col < 196) {
      _Float16* out = u2 + ((long)bt * 196 + col) * 32 + mtile * 16 + roff;
#pragma unroll
      for (int r = 0; r < 8; ++r) out[r] = (_Float16)acc[r];
    }
  }
}

// ---------------------------------------------------------------------------
// pool + psp + spike, layer 4.  Produces dense-layer B layout:
// s4[(t*32 + b)*1568 + k], k = c*49 + h2*7 + w2 (reference flatten order).
// ---------------------------------------------------------------------------
__global__ void pool_psp2_kernel(const _Float16* __restrict__ s3,
                                 _Float16* __restrict__ s4) {
  int tid = blockIdx.x * 256 + threadIdx.x;
  if (tid >= 32 * 1568) return;
  int k = tid % 1568;
  int b = tid / 1568;
  int c = k / 49;
  int r = k % 49;
  int h2 = r / 7, w2 = r % 7;
  int s00 = (h2 * 2) * 14 + w2 * 2;
  float p = 0.0f, q = 0.0f;
  for (int t = 0; t < T_SIM; ++t) {
    long base = ((long)(b * T_SIM + t) * 196) * 32 + c;
    float x = (float)s3[base + s00 * 32] + (float)s3[base + (s00 + 1) * 32]
            + (float)s3[base + (s00 + 14) * 32] + (float)s3[base + (s00 + 15) * 32];
    x *= 1.1f;
    p = ALPHA * p + x;
    q = ALPHA * q + p;
    float y = PSPC * (q - p);
    s4[((long)t * 32 + b) * 1568 + k] = (_Float16)((y >= 1.0f) ? 1.0f : 0.0f);
  }
}

// ---------------------------------------------------------------------------
// dense1: Y[410x32] = Wf1[410x1568] * X[1568x32] per t.
// grid (100, 13), 4 waves/block, one output tile per wave (52 tiles).
// B-fragments: contiguous 32B loads from s4[t][b][K].
// u5 column-major f16: u5[(t*32 + b)*416 + o].
// ---------------------------------------------------------------------------
__global__ void __launch_bounds__(128)
dense1_kernel(const _Float16* __restrict__ s4, const _Float16* __restrict__ Apk,
              _Float16* __restrict__ u5) {
  int t  = blockIdx.x;
  int g  = blockIdx.y;
  int lane = threadIdx.x & 31;
  int wv   = threadIdx.x >> 5;
  int tile = g * 4 + wv;           // 0..51
  int mtile = tile >> 1;           // 0..25
  int ntile = tile & 1;
  int col  = ntile * 16 + (lane & 15);
  int koff = (lane < 16) ? 0 : 16;
  int roff = (lane < 16) ? 0 : 8;
  const _Float16* bp = s4 + ((long)t * 32 + col) * 1568 + koff;
  const _Float16* ap = Apk + ((long)mtile * 49 * 32 + lane) * 16;
  v8f acc = {};
  for (int ks = 0; ks < 49; ++ks) {
    v16h a   = *(const v16h*)(ap + (long)ks * 512);
    v16h bfr = *(const v16h*)(bp + ks * 32);
    __builtin_prefetch(ap + (long)(ks + 1) * 512, 0, 0);
    acc = wmma_f16(a, bfr, acc);
  }
  _Float16* out = u5 + ((long)t * 32 + col) * 416 + mtile * 16 + roff;
#pragma unroll
  for (int r = 0; r < 8; ++r) out[r] = (_Float16)acc[r];
}

// ---------------------------------------------------------------------------
// dense2: Y[10(->16)x32] = Wf2 * X[410(->416)x32] per t.  1 M-tile, 2 N-tiles.
// u6 f16: u6[(t*32 + b)*16 + o].
// ---------------------------------------------------------------------------
__global__ void __launch_bounds__(64)
dense2_kernel(const _Float16* __restrict__ s5, const _Float16* __restrict__ Apk,
              _Float16* __restrict__ u6) {
  int t = blockIdx.x;
  int lane  = threadIdx.x & 31;
  int ntile = threadIdx.x >> 5;
  int col  = ntile * 16 + (lane & 15);
  int koff = (lane < 16) ? 0 : 16;
  int roff = (lane < 16) ? 0 : 8;
  const _Float16* bp = s5 + ((long)t * 32 + col) * 416 + koff;
  v8f acc = {};
#pragma unroll
  for (int ks = 0; ks < 13; ++ks) {
    v16h a   = *(const v16h*)(Apk + (ks * 32 + lane) * 16);
    v16h bfr = *(const v16h*)(bp + ks * 32);
    acc = wmma_f16(a, bfr, acc);
  }
  _Float16* out = u6 + ((long)t * 32 + col) * 16 + roff;
#pragma unroll
  for (int r = 0; r < 8; ++r) out[r] = (_Float16)acc[r];
}

// ---------------------------------------------------------------------------
// Final psp + spike, writes f32 output in reference layout [b][o][t].
// ---------------------------------------------------------------------------
__global__ void psp_out_kernel(const _Float16* __restrict__ u6,
                               float* __restrict__ out) {
  int tid = blockIdx.x * 64 + threadIdx.x;
  if (tid >= 320) return;
  int o = tid % 10;
  int b = tid / 10;
  float p = 0.0f, q = 0.0f;
  for (int t = 0; t < T_SIM; ++t) {
    float x = (float)u6[(t * 32 + b) * 16 + o];
    p = ALPHA * p + x;
    q = ALPHA * q + p;
    float y = PSPC * (q - p);
    out[(b * 10 + o) * T_SIM + t] = (y >= 1.0f) ? 1.0f : 0.0f;
  }
}

// ---------------------------------------------------------------------------
extern "C" void kernel_launch(void* const* d_in, const int* in_sizes, int n_in,
                              void* d_out, int out_size, void* d_ws, size_t ws_size,
                              hipStream_t stream) {
  const float* img = (const float*)d_in[0];
  const float* ru  = (const float*)d_in[1];
  const float* w1  = (const float*)d_in[2];
  const float* w2  = (const float*)d_in[3];
  const float* wf1 = (const float*)d_in[4];
  const float* wf2 = (const float*)d_in[5];
  float* out = (float*)d_out;

  // Workspace arena (offsets aligned to 256B); buffers reused across layers.
  char* ws = (char*)d_ws;
  size_t off = 0;
  auto alloc = [&](size_t bytes) -> char* {
    char* p = ws + off;
    off = (off + bytes + 255) & ~(size_t)255;
    return p;
  };
  _Float16* apk1  = (_Float16*)alloc((size_t)1 * 1 * 512 * 2);
  _Float16* apk2  = (_Float16*)alloc((size_t)2 * 13 * 512 * 2);
  _Float16* apkf1 = (_Float16*)alloc((size_t)26 * 49 * 512 * 2);
  _Float16* apkf2 = (_Float16*)alloc((size_t)1 * 13 * 512 * 2);
  _Float16* spk0  = (_Float16*)alloc((size_t)3200 * 784 * 2);          //  5.0 MB
  _Float16* ub    = (_Float16*)alloc((size_t)3200 * 784 * 16 * 2);     //  80 MB (u1/u2/u5/u6)
  _Float16* sa    = (_Float16*)alloc((size_t)3200 * 784 * 16 * 2);     //  80 MB (s1/s3/s5)
  _Float16* sb    = (_Float16*)alloc((size_t)3200 * 196 * 16 * 2);     //  20 MB (s2/s4)

  // Pack weights into WMMA A-fragment layouts.
  pack_a_kernel<<<(512 + 255) / 256, 256, 0, stream>>>(w1, apk1, 16, 25, 1, 1, 0);
  pack_a_kernel<<<(2 * 13 * 512 + 255) / 256, 256, 0, stream>>>(w2, apk2, 32, 416, 2, 13, 1);
  pack_a_kernel<<<(26 * 49 * 512 + 255) / 256, 256, 0, stream>>>(wf1, apkf1, 410, 1568, 26, 49, 0);
  pack_a_kernel<<<(13 * 512 + 255) / 256, 256, 0, stream>>>(wf2, apkf2, 10, 410, 1, 13, 0);

  // Rate encoding.
  encode_kernel<<<(32 * 784 * T_SIM + 255) / 256, 256, 0, stream>>>(img, ru, spk0);

  // Layer 1: conv 1->16 + psp + spike.
  conv1_kernel<<<3200, 128, 0, stream>>>(spk0, apk1, ub);
  psp_spike_kernel<<<(32 * 12544 + 255) / 256, 256, 0, stream>>>(ub, sa, 32 * 12544, 12544, 1);

  // Layer 2: pool + psp + spike.
  pool_psp1_kernel<<<(32 * 196 * 16 + 255) / 256, 256, 0, stream>>>(sa, sb);

  // Layer 3: conv 16->32 + psp + spike.
  conv2_kernel<<<3200, 256, 0, stream>>>(sb, apk2, ub);
  psp_spike_kernel<<<(32 * 6272 + 255) / 256, 256, 0, stream>>>(ub, sa, 32 * 6272, 6272, 1);

  // Layer 4: pool + psp + spike (-> dense layout).
  pool_psp2_kernel<<<(32 * 1568 + 255) / 256, 256, 0, stream>>>(sa, sb);

  // Layer 5: dense 1568->410 + psp + spike.
  dim3 g1(T_SIM, 13);
  dense1_kernel<<<g1, 128, 0, stream>>>(sb, apkf1, ub);
  psp_spike_kernel<<<(32 * 416 + 255) / 256, 256, 0, stream>>>(ub, sa, 32 * 416, 32 * 416, 0);

  // Layer 6: dense 410->10 + psp + spike -> output.
  dense2_kernel<<<T_SIM, 64, 0, stream>>>(sa, apkf2, ub);
  psp_out_kernel<<<(320 + 63) / 64, 64, 0, stream>>>(ub, out);
}
